// LSTM2_74354473828953
// MI455X (gfx1250) — matrix operand order repaired and checked by reading, not compiled, not run on hardware
//
#include <hip/hip_runtime.h>
#include <hip/hip_bf16.h>

#define SEQ    512
#define BATCH  256
#define INDIM  15
#define H      512
#define G4     2048
#define FC_HID 64
#define NBLK   64
#define NTHR   256

typedef unsigned short ushort_t;
typedef __attribute__((ext_vector_type(16))) __bf16    v16bf;
typedef __attribute__((ext_vector_type(8)))  float     v8f;
typedef __attribute__((ext_vector_type(4)))  unsigned  uint4v;

union BV { uint4v u[2]; v16bf v; };

// ---------- helpers ----------

__device__ __forceinline__ ushort_t f2bf(float f) {
  unsigned u = __float_as_uint(f);
  u += 0x7FFFu + ((u >> 16) & 1u);      // round-to-nearest-even
  return (ushort_t)(u >> 16);
}

__device__ __forceinline__ float sigmoidf_(float x) {
  return 1.0f / (1.0f + __expf(-x));
}

// A fragment: 16x32 bf16, row-major source, stride in elements.
// lane L: row m = L%16; sel = L/16 selects K 0..15 / 16..31
__device__ __forceinline__ v16bf load_a(const ushort_t* __restrict__ p,
                                        int row, int stride, int kbase, int sel) {
  BV r;
  const ushort_t* base = p + (size_t)row * stride + kbase + sel * 16;
  r.u[0] = *reinterpret_cast<const uint4v*>(base);
  r.u[1] = *reinterpret_cast<const uint4v*>(base + 8);
  return r.v;
}

// B fragment: 32x16 bf16 with B[k][n] = W[col_n][k]; W row-major, stride elems.
// lane L: column n = L%16; khalf = L/16 selects K 0..15 / 16..31.
__device__ __forceinline__ v16bf load_b(const ushort_t* __restrict__ p,
                                        int col, int stride, int kbase, int khalf) {
  BV r;
  const uint4v* q = reinterpret_cast<const uint4v*>(p + (size_t)col * stride + kbase + khalf * 16);
  r.u[0] = q[0];
  r.u[1] = q[1];
  return r.v;
}

__device__ __forceinline__ v8f wmma_bf16(v16bf a, v16bf b, v8f c) {
  return __builtin_amdgcn_wmma_f32_16x16x32_bf16(
      /*neg_a=*/false, a, /*neg_b=*/false, b,
      /*c_mod=*/(short)0, c, /*reuse_a=*/false, /*reuse_b=*/false);
}

// acc[g] += A(16-row tile, K = KB*32) x B(W rows g*H + j0 ..)
// Unroll-by-2 software pipeline with two alternating buffer sets:
// no cross-iteration register copies, one k-block of loads always in flight.
template <int KB>
__device__ __forceinline__ void gemm_tiles(const ushort_t* __restrict__ Abuf, int arow, int astride,
                                           const ushort_t* __restrict__ Bbuf, int j0, int bstride,
                                           v8f acc[4], int sel) {
  v16bf a0 = load_a(Abuf, arow, astride, 0, sel);
  v16bf b0[4];
#pragma unroll
  for (int g = 0; g < 4; ++g) b0[g] = load_b(Bbuf, g * H + j0, bstride, 0, sel);

  int kb = 0;
#pragma unroll 1
  for (; kb + 2 < KB; kb += 2) {
    v16bf a1 = load_a(Abuf, arow, astride, (kb + 1) * 32, sel);
    v16bf b1[4];
#pragma unroll
    for (int g = 0; g < 4; ++g) b1[g] = load_b(Bbuf, g * H + j0, bstride, (kb + 1) * 32, sel);
#pragma unroll
    for (int g = 0; g < 4; ++g) acc[g] = wmma_bf16(a0, b0[g], acc[g]);

    a0 = load_a(Abuf, arow, astride, (kb + 2) * 32, sel);   // reload buf0 in place
#pragma unroll
    for (int g = 0; g < 4; ++g) b0[g] = load_b(Bbuf, g * H + j0, bstride, (kb + 2) * 32, sel);
#pragma unroll
    for (int g = 0; g < 4; ++g) acc[g] = wmma_bf16(a1, b1[g], acc[g]);
  }

  if (kb + 1 < KB) {   // even KB: final pair (compile-time resolved)
    v16bf a1 = load_a(Abuf, arow, astride, (kb + 1) * 32, sel);
    v16bf b1[4];
#pragma unroll
    for (int g = 0; g < 4; ++g) b1[g] = load_b(Bbuf, g * H + j0, bstride, (kb + 1) * 32, sel);
#pragma unroll
    for (int g = 0; g < 4; ++g) acc[g] = wmma_bf16(a0, b0[g], acc[g]);
#pragma unroll
    for (int g = 0; g < 4; ++g) acc[g] = wmma_bf16(a1, b1[g], acc[g]);
  } else {             // KB == 1
#pragma unroll
    for (int g = 0; g < 4; ++g) acc[g] = wmma_bf16(a0, b0[g], acc[g]);
  }
}

// Device-wide generation spin barrier (all NBLK blocks co-resident).
__device__ __forceinline__ void grid_sync(unsigned* cnt, unsigned* gen) {
  __syncthreads();
  if (threadIdx.x == 0) {
    unsigned g = __hip_atomic_load(gen, __ATOMIC_RELAXED, __HIP_MEMORY_SCOPE_AGENT);
    unsigned prev = __hip_atomic_fetch_add(cnt, 1u, __ATOMIC_ACQ_REL, __HIP_MEMORY_SCOPE_AGENT);
    if (prev == NBLK - 1) {
      __hip_atomic_store(cnt, 0u, __ATOMIC_RELAXED, __HIP_MEMORY_SCOPE_AGENT);
      __hip_atomic_fetch_add(gen, 1u, __ATOMIC_ACQ_REL, __HIP_MEMORY_SCOPE_AGENT);
    } else {
      while (__hip_atomic_load(gen, __ATOMIC_ACQUIRE, __HIP_MEMORY_SCOPE_AGENT) == g) {
        __builtin_amdgcn_s_sleep(4);
      }
    }
  }
  __syncthreads();
}

// Pointwise LSTM cell update, fully on the 4 gate accumulators.
// C/D layout: lane L -> n = L%16, VGPR r -> m = (L/16)*8 + r.
__device__ __forceinline__ void cell_update(const v8f acc[4],
                                            float* __restrict__ cbuf,
                                            ushort_t* __restrict__ hbf,
                                            float* __restrict__ hf,
                                            int mt, int jt, int lane) {
  const int nl = lane & 15;
  const int mh = lane >> 4;
  const int j  = jt * 16 + nl;
#pragma unroll
  for (int r = 0; r < 8; ++r) {
    const int m = mt * 16 + mh * 8 + r;
    const size_t idx = (size_t)m * H + j;
    float ig = sigmoidf_(acc[0][r]);
    float fg = sigmoidf_(acc[1][r]);
    float gg = tanhf(acc[2][r]);
    float og = sigmoidf_(acc[3][r]);
    float cn = fg * cbuf[idx] + ig * gg;
    float hn = og * tanhf(cn);
    cbuf[idx] = cn;
    hbf[idx]  = f2bf(hn);
    if (hf) hf[idx] = hn;
  }
}

// ---------- kernels ----------

__global__ void init_kernel(float* c1, float* c2, ushort_t* h1bf, ushort_t* h2bf,
                            float* h2f, float* bias1, float* bias2,
                            const float* __restrict__ bih1, const float* __restrict__ bhh1,
                            const float* __restrict__ bih2, const float* __restrict__ bhh2,
                            unsigned* bar) {
  int i = blockIdx.x * blockDim.x + threadIdx.x;
  int stride = gridDim.x * blockDim.x;
  for (int k = i; k < BATCH * H; k += stride) {
    c1[k] = 0.f; c2[k] = 0.f; h2f[k] = 0.f;
    h1bf[k] = 0; h2bf[k] = 0;
  }
  for (int k = i; k < G4; k += stride) {
    bias1[k] = bih1[k] + bhh1[k];
    bias2[k] = bih2[k] + bhh2[k];
  }
  if (i == 0) { bar[0] = 0u; bar[1] = 0u; }
}

// dst[r*dcols + c] = c < scols ? bf16(src[r*scols + c]) : 0
__global__ void conv_bf16_pad(const float* __restrict__ src, ushort_t* __restrict__ dst,
                              int scols, int dcols, long total) {
  long i = (long)blockIdx.x * blockDim.x + threadIdx.x;
  long stride = (long)gridDim.x * blockDim.x;
  for (; i < total; i += stride) {
    long r = i / dcols;
    int  c = (int)(i - r * dcols);
    dst[i] = (c < scols) ? f2bf(src[r * scols + c]) : (ushort_t)0;
  }
}

__global__ void __launch_bounds__(NTHR)
lstm_persistent(const ushort_t* __restrict__ xbf,    // [SEQ][BATCH][32] bf16, K-padded
                const ushort_t* __restrict__ Wih1p,  // [G4][32]  bf16, K-padded
                const ushort_t* __restrict__ Whh1,   // [G4][H]   bf16
                const ushort_t* __restrict__ Wih2,   // [G4][H]   bf16
                const ushort_t* __restrict__ Whh2,   // [G4][H]   bf16
                const float* __restrict__ bias1,     // [G4] = b_ih1 + b_hh1
                const float* __restrict__ bias2,     // [G4]
                float* __restrict__ c1, float* __restrict__ c2,
                ushort_t* __restrict__ h1bf, ushort_t* __restrict__ h2bf,
                float* __restrict__ h2f,
                unsigned* bar) {
  const int lane = threadIdx.x & 31;
  const int wave = threadIdx.x >> 5;                 // 0..7
  // All 8 waves of a block share jt -> identical weight (B) lines served by the
  // read-only WGP$; A (h-tile) fragments differ per wave but are 4x smaller.
  const int jt   = blockIdx.x & 31;                  // hidden-tile (32 x 16 over H)
  const int mt   = (blockIdx.x >> 5) * 8 + wave;     // batch-tile  (16 x 16 over BATCH)
  const int nl   = lane & 15;
  const int sel  = lane >> 4; // k-group / k-half / m-half selector
  const int j0   = jt * 16 + nl;
  const int arow = mt * 16 + nl;
  unsigned* cnt = bar;
  unsigned* gen = bar + 1;

#pragma unroll 1
  for (int t = 0; t < SEQ; ++t) {
    // ================= layer 1 =================
    v8f acc[4];
#pragma unroll
    for (int g = 0; g < 4; ++g) {
      float bv = bias1[g * H + j0];
#pragma unroll
      for (int r = 0; r < 8; ++r) acc[g][r] = bv;
    }
    // x_t @ W_ih1^T : one K=32 (zero-padded) block
    gemm_tiles<1>(xbf + (size_t)t * BATCH * 32, arow, 32, Wih1p, j0, 32, acc, sel);
    // h1 @ W_hh1^T
    gemm_tiles<H / 32>(h1bf, arow, H, Whh1, j0, H, acc, sel);
    cell_update(acc, c1, h1bf, nullptr, mt, jt, lane);  // only own tiles touched
    grid_sync(cnt, gen);                                // publish h1

    // ================= layer 2 =================
    v8f acc2[4];
#pragma unroll
    for (int g = 0; g < 4; ++g) {
      float bv = bias2[g * H + j0];
#pragma unroll
      for (int r = 0; r < 8; ++r) acc2[g][r] = bv;
    }
    gemm_tiles<H / 32>(h1bf, arow, H, Wih2, j0, H, acc2, sel);   // h1 @ W_ih2^T
    gemm_tiles<H / 32>(h2bf, arow, H, Whh2, j0, H, acc2, sel);   // h2 @ W_hh2^T
    cell_update(acc2, c2, h2bf, h2f, mt, jt, lane);
    grid_sync(cnt, gen);                                // publish h2
  }
}

// Tiny head: one thread per batch row, fp32 throughout (16.8 MFLOP total).
__global__ void head_kernel(const float* __restrict__ h2f,
                            const float* __restrict__ fc1w, const float* __restrict__ fc1b,
                            const float* __restrict__ fcw,  const float* __restrict__ fcb,
                            float* __restrict__ out) {
  int b = threadIdx.x;  // 256 threads
  float o = fcb[0];
  for (int u = 0; u < FC_HID; ++u) {
    float s = fc1b[u];
    for (int k = 0; k < H; ++k) {
      float hv = h2f[(size_t)b * H + k];
      hv = hv > 0.f ? hv : 0.f;
      s += hv * fc1w[u * H + k];
    }
    s = s > 0.f ? s : 0.f;
    o += s * fcw[u];
  }
  out[b] = 2.0f * o;  // reference returns out + out
}

// ---------- host side ----------

extern "C" void kernel_launch(void* const* d_in, const int* in_sizes, int n_in,
                              void* d_out, int out_size, void* d_ws, size_t ws_size,
                              hipStream_t stream) {
  (void)in_sizes; (void)n_in; (void)out_size; (void)ws_size;
  const float* x    = (const float*)d_in[0];
  const float* Wih1 = (const float*)d_in[1];
  const float* Whh1 = (const float*)d_in[2];
  const float* bih1 = (const float*)d_in[3];
  const float* bhh1 = (const float*)d_in[4];
  const float* Wih2 = (const float*)d_in[5];
  const float* Whh2 = (const float*)d_in[6];
  const float* bih2 = (const float*)d_in[7];
  const float* bhh2 = (const float*)d_in[8];
  const float* fc1w = (const float*)d_in[9];
  const float* fc1b = (const float*)d_in[10];
  const float* fcw  = (const float*)d_in[11];
  const float* fcb  = (const float*)d_in[12];
  float* out = (float*)d_out;

  char* ws = (char*)d_ws;
  size_t off = 0;
  auto alloc = [&](size_t bytes) -> char* {
    off = (off + 255) & ~(size_t)255;
    char* p = ws + off;
    off += bytes;
    return p;
  };

  unsigned* bar    = (unsigned*)alloc(256);
  float*    bias1  = (float*)   alloc((size_t)G4 * 4);
  float*    bias2  = (float*)   alloc((size_t)G4 * 4);
  ushort_t* Wih1p  = (ushort_t*)alloc((size_t)G4 * 32 * 2);
  ushort_t* Whh1b  = (ushort_t*)alloc((size_t)G4 * H * 2);
  ushort_t* Wih2b  = (ushort_t*)alloc((size_t)G4 * H * 2);
  ushort_t* Whh2b  = (ushort_t*)alloc((size_t)G4 * H * 2);
  ushort_t* xbf    = (ushort_t*)alloc((size_t)SEQ * BATCH * 32 * 2);
  ushort_t* h1bf   = (ushort_t*)alloc((size_t)BATCH * H * 2);
  ushort_t* h2bf   = (ushort_t*)alloc((size_t)BATCH * H * 2);
  float*    c1     = (float*)   alloc((size_t)BATCH * H * 4);
  float*    c2     = (float*)   alloc((size_t)BATCH * H * 4);
  float*    h2f    = (float*)   alloc((size_t)BATCH * H * 4);

  init_kernel<<<512, 256, 0, stream>>>(c1, c2, h1bf, h2bf, h2f, bias1, bias2,
                                       bih1, bhh1, bih2, bhh2, bar);

  conv_bf16_pad<<<1024, 256, 0, stream>>>(Whh1, Whh1b, H, H, (long)G4 * H);
  conv_bf16_pad<<<1024, 256, 0, stream>>>(Wih2, Wih2b, H, H, (long)G4 * H);
  conv_bf16_pad<<<1024, 256, 0, stream>>>(Whh2, Whh2b, H, H, (long)G4 * H);
  conv_bf16_pad<<<128,  256, 0, stream>>>(Wih1, Wih1p, INDIM, 32, (long)G4 * 32);
  conv_bf16_pad<<<2048, 256, 0, stream>>>(x, xbf, INDIM, 32, (long)SEQ * BATCH * 32);

  lstm_persistent<<<NBLK, NTHR, 0, stream>>>(xbf, Wih1p, Whh1b, Wih2b, Whh2b,
                                             bias1, bias2, c1, c2, h1bf, h2bf,
                                             h2f, bar);

  head_kernel<<<1, 256, 0, stream>>>(h2f, fc1w, fc1b, fcw, fcb, out);
}
